// MambaPlusPlus_layer_48103633715536
// MI455X (gfx1250) — matrix operand order repaired and checked by reading, not compiled, not run on hardware
//
#include <hip/hip_runtime.h>

typedef __attribute__((ext_vector_type(16))) __bf16 v16bf;
typedef __attribute__((ext_vector_type(8)))  __bf16 v8bf;
typedef __attribute__((ext_vector_type(8)))  float  v8f;

constexpr int EMBED  = 1024;
constexpr int HID    = 1024;
constexpr int FF     = 4096;
constexpr int DH     = 64;
constexpr int BB     = 4;
constexpr int LL     = 4096;
constexpr int NTOK   = BB * LL;     // 16384
constexpr int CHAINS = BB * HID;    // 4096
constexpr int CHUNK  = 512;
constexpr int NCHUNK = LL / CHUNK;  // 8

constexpr int BSTRIDE  = 40;            // LDS row stride (bf16 elems): 80B, padded
constexpr int BUFELEMS = 64 * BSTRIDE;  // one staged 64(N) x 32(K) weight tile

// ---------------------------------------------------------------------------
// Wait helpers for the async-to-LDS pipeline
// ---------------------------------------------------------------------------
__device__ __forceinline__ void wait_async_le1() {
#if __has_builtin(__builtin_amdgcn_s_wait_asynccnt)
  __builtin_amdgcn_s_wait_asynccnt(1);
#else
  asm volatile("s_wait_asynccnt 0x1" ::: "memory");
#endif
}
__device__ __forceinline__ void wait_async_le3() {
#if __has_builtin(__builtin_amdgcn_s_wait_asynccnt)
  __builtin_amdgcn_s_wait_asynccnt(3);
#else
  asm volatile("s_wait_asynccnt 0x3" ::: "memory");
#endif
}
__device__ __forceinline__ void wait_async_0() {
#if __has_builtin(__builtin_amdgcn_s_wait_asynccnt)
  __builtin_amdgcn_s_wait_asynccnt(0);
#else
  asm volatile("s_wait_asynccnt 0x0" ::: "memory");
#endif
}

// Async copy of one 64(N) x 32(K) bf16 weight tile into LDS.
// 256 threads: thread -> (row = tid/4, 16B quarter = tid%4). One
// global_load_async_to_lds_b128 per thread (1 ASYNCcnt tick per wave).
__device__ __forceinline__ void async_copy_Btile(const __bf16* __restrict__ W, int ldw,
                                                 int nBase, int kb, __bf16* sbuf,
                                                 int tid) {
  const int row = tid >> 2;
  const int q   = tid & 3;
  const __bf16* src = W + (size_t)(nBase + row) * ldw + kb + q * 8;
  unsigned lds = (unsigned)(uintptr_t)(sbuf + row * BSTRIDE + q * 8);
  asm volatile("global_load_async_to_lds_b128 %0, %1, off"
               :: "v"(lds), "v"(src) : "memory");
}

// ---------------------------------------------------------------------------
// WMMA fragment loaders (bf16 operands, per CDNA5 ISA 16-bit layouts)
// A (16x32 MxK): lane 0-15 = M rows, K 0..7 & 16..23; lanes 16-31: K 8..15 & 24..31
// B (32x16 KxN): lane n = column n; lanes 0-15 K 0..7 & 16..23; lanes 16-31: K 8..15 & 24..31
// ---------------------------------------------------------------------------
__device__ __forceinline__ v16bf load_A_frag(const __bf16* __restrict__ X, int ldx,
                                             int mBase, int kBase, int lane) {
  const int m  = lane & 15;
  const int hk = lane >> 4;
  const __bf16* p0 = X + (size_t)(mBase + m) * ldx + kBase + 8 * hk;
  v8bf lo = *(const v8bf*)p0;          // K = kBase + 8*hk + 0..7
  v8bf hi = *(const v8bf*)(p0 + 16);   // K = kBase + 16 + 8*hk + 0..7
  return __builtin_shufflevector(lo, hi, 0, 1, 2, 3, 4, 5, 6, 7,
                                 8, 9, 10, 11, 12, 13, 14, 15);
}

__device__ __forceinline__ v16bf load_B_frag_lds(const __bf16* sbuf, int nOff, int lane) {
  const int n  = lane & 15;
  const int hk = lane >> 4;
  const __bf16* p = sbuf + (nOff + n) * BSTRIDE + 8 * hk;
  v8bf lo = *(const v8bf*)p;
  v8bf hi = *(const v8bf*)(p + 16);
  return __builtin_shufflevector(lo, hi, 0, 1, 2, 3, 4, 5, 6, 7,
                                 8, 9, 10, 11, 12, 13, 14, 15);
}

__device__ __forceinline__ v8f wmma_bf16(v16bf a, v16bf b, v8f c) {
  return __builtin_amdgcn_wmma_f32_16x16x32_bf16(false, a, false, b, (short)0,
                                                 c, false, false);
}

// ---------------------------------------------------------------------------
// Block-cooperative GEMM core: block = 8 waves; wave w computes 16x64 at
// mBase = mBlock*128 + w*16. Weight tile double-buffered in LDS via async DMA.
// ---------------------------------------------------------------------------
__device__ __forceinline__ void coop_gemm(const __bf16* __restrict__ A, int lda,
                                          const __bf16* __restrict__ W, int ldw,
                                          int mBase, int nBase, int K,
                                          __bf16* sB, v8f* acc) {
  const int tid  = threadIdx.x;
  const int lane = tid & 31;
  async_copy_Btile(W, ldw, nBase, 0, sB, tid);
  for (int kb = 0; kb < K; kb += 32) {
    __bf16* cur = sB + ((kb >> 5) & 1) * BUFELEMS;
    __bf16* nxt = sB + (((kb >> 5) + 1) & 1) * BUFELEMS;
    if (kb + 32 < K) {
      async_copy_Btile(W, ldw, nBase, kb + 32, nxt, tid);
      wait_async_le1();              // previous tile landed; next in flight
    } else {
      wait_async_0();
    }
    __syncthreads();
    v16bf Af = load_A_frag(A, lda, mBase, kb, lane);
#pragma unroll
    for (int t = 0; t < 4; ++t) {
      v16bf Bf = load_B_frag_lds(cur, 16 * t, lane);
      acc[t] = wmma_bf16(Af, Bf, acc[t]);
    }
    __syncthreads();                 // done reading cur before it is re-filled
  }
}

// ---------------------------------------------------------------------------
// fp32 -> bf16 pack (4 elems/thread)
// ---------------------------------------------------------------------------
__global__ void k_pack4(const float* __restrict__ s, __bf16* __restrict__ d, int n) {
  const int i = (blockIdx.x * blockDim.x + threadIdx.x) * 4;
  if (i + 3 < n) {
    const float4 v = *(const float4*)(s + i);
    d[i + 0] = (__bf16)v.x;
    d[i + 1] = (__bf16)v.y;
    d[i + 2] = (__bf16)v.z;
    d[i + 3] = (__bf16)v.w;
  }
}

// ---------------------------------------------------------------------------
// 1) a = tanh(emb@Wa^T+ba); b = emb@Wb^T+bb; w = emb@Wo^T+bo
//    Block-cooperative: 128x64 tile for all three projections; three weight
//    tiles staged per K-step via async DMA; A fragment reused by 12 WMMAs.
// ---------------------------------------------------------------------------
__global__ void k_abw(const __bf16* __restrict__ emb,
                      const __bf16* __restrict__ Wa, const float* __restrict__ ba,
                      const __bf16* __restrict__ Wb, const float* __restrict__ bbv,
                      const __bf16* __restrict__ Wo, const float* __restrict__ bo,
                      float* __restrict__ a, float* __restrict__ b,
                      float* __restrict__ w) {
  __shared__ __bf16 sB[2][3][BUFELEMS];   // 30 KB
  const int ng    = blockIdx.x & 15;      // HID/64 = 16 n-groups
  const int mblk  = blockIdx.x >> 4;
  const int mBase = mblk * 128 + (threadIdx.x >> 5) * 16;
  const int nBase = ng * 64;
  const int tid   = threadIdx.x;
  const int lane  = tid & 31;

  v8f accA[4] = {}, accB[4] = {}, accW[4] = {};

  async_copy_Btile(Wa, EMBED, nBase, 0, sB[0][0], tid);
  async_copy_Btile(Wb, EMBED, nBase, 0, sB[0][1], tid);
  async_copy_Btile(Wo, EMBED, nBase, 0, sB[0][2], tid);
  for (int kb = 0; kb < EMBED; kb += 32) {
    const int cur = (kb >> 5) & 1, nxt = cur ^ 1;
    if (kb + 32 < EMBED) {
      async_copy_Btile(Wa, EMBED, nBase, kb + 32, sB[nxt][0], tid);
      async_copy_Btile(Wb, EMBED, nBase, kb + 32, sB[nxt][1], tid);
      async_copy_Btile(Wo, EMBED, nBase, kb + 32, sB[nxt][2], tid);
      wait_async_le3();              // previous 3 tiles landed; next 3 in flight
    } else {
      wait_async_0();
    }
    __syncthreads();
    v16bf Af = load_A_frag(emb, EMBED, mBase, kb, lane);
#pragma unroll
    for (int t = 0; t < 4; ++t) {
      accA[t] = wmma_bf16(Af, load_B_frag_lds(sB[cur][0], 16 * t, lane), accA[t]);
      accB[t] = wmma_bf16(Af, load_B_frag_lds(sB[cur][1], 16 * t, lane), accB[t]);
      accW[t] = wmma_bf16(Af, load_B_frag_lds(sB[cur][2], 16 * t, lane), accW[t]);
    }
    __syncthreads();
  }
  const int n = lane & 15, hk = lane >> 4;
#pragma unroll
  for (int t = 0; t < 4; ++t) {
    const int col = nBase + 16 * t + n;
    const float biasA = ba[col], biasB = bbv[col], biasW = bo[col];
#pragma unroll
    for (int j = 0; j < 8; ++j) {
      const int row = mBase + j + 8 * hk;
      const size_t idx = (size_t)row * HID + col;
      a[idx] = tanhf(accA[t][j] + biasA);
      b[idx] = accB[t][j] + biasB;
      w[idx] = accW[t][j] + biasW;
    }
  }
}

// ---------------------------------------------------------------------------
// 2-4) Chunked associative scan: h_t = a_t*h_{t-1} + b_t, h_0 = 0.
// ---------------------------------------------------------------------------
__global__ void k_scan_local(const float* __restrict__ a, const float* __restrict__ b,
                             float* __restrict__ Aprod, float* __restrict__ Bfin) {
  const int tid = blockIdx.x * blockDim.x + threadIdx.x;
  const int chain = tid & (CHAINS - 1);
  const int chunk = tid / CHAINS;
  const int batch = chain >> 10;
  const int c     = chain & (HID - 1);
  size_t base = ((size_t)batch * LL + (size_t)chunk * CHUNK) * HID + c;
  float A = 1.f, Bv = 0.f;
  for (int t = 0; t < CHUNK; ++t) {
    const float av = a[base], bv = b[base];
    Bv = av * Bv + bv;
    A *= av;
    base += HID;
  }
  Aprod[chunk * CHAINS + chain] = A;
  Bfin[chunk * CHAINS + chain]  = Bv;
}

__global__ void k_scan_carry(const float* __restrict__ Aprod,
                             const float* __restrict__ Bfin,
                             float* __restrict__ hstart) {
  const int chain = blockIdx.x * blockDim.x + threadIdx.x;
  float hs = 0.f;
#pragma unroll
  for (int ck = 0; ck < NCHUNK; ++ck) {
    hstart[ck * CHAINS + chain] = hs;
    hs = Aprod[ck * CHAINS + chain] * hs + Bfin[ck * CHAINS + chain];
  }
}

__global__ void k_scan_apply(const float* __restrict__ a, const float* __restrict__ b,
                             const float* __restrict__ hstart,
                             __bf16* __restrict__ h16) {
  const int tid = blockIdx.x * blockDim.x + threadIdx.x;
  const int chain = tid & (CHAINS - 1);
  const int chunk = tid / CHAINS;
  const int batch = chain >> 10;
  const int c     = chain & (HID - 1);
  size_t base = ((size_t)batch * LL + (size_t)chunk * CHUNK) * HID + c;
  float h = hstart[chunk * CHAINS + chain];
  for (int t = 0; t < CHUNK; ++t) {
    h = a[base] * h + b[base];
    h16[base] = (__bf16)h;           // feeds the C-projection GEMM
    base += HID;
  }
}

// ---------------------------------------------------------------------------
// 5) c = h@C^T + Cb; z = head_w[head]*(c + w). Block tile 128x64 = one head.
// ---------------------------------------------------------------------------
__global__ void k_cz(const __bf16* __restrict__ h, const __bf16* __restrict__ Cw,
                     const float* __restrict__ Cb, const float* __restrict__ wbuf,
                     const float* __restrict__ headw, float* __restrict__ z) {
  __shared__ __bf16 sB[2 * BUFELEMS];
  const int ng    = blockIdx.x & 15;          // 16 n-groups of 64
  const int mblk  = blockIdx.x >> 4;
  const int mBase = mblk * 128 + (threadIdx.x >> 5) * 16;
  const int nBase = ng * DH;
  const int lane  = threadIdx.x & 31;

  v8f acc[4] = {};
  coop_gemm(h, HID, Cw, HID, mBase, nBase, HID, sB, acc);

  const float hw = headw[ng];
  const int n = lane & 15, hk = lane >> 4;
#pragma unroll
  for (int t = 0; t < 4; ++t) {
    const int col = nBase + 16 * t + n;
    const float bias = Cb[col];
#pragma unroll
    for (int j = 0; j < 8; ++j) {
      const int row = mBase + j + 8 * hk;
      const size_t idx = (size_t)row * HID + col;
      z[idx] = hw * (acc[t][j] + bias + wbuf[idx]);
    }
  }
}

// ---------------------------------------------------------------------------
// 6) u = z + layernorm(z). Writes fp32 (residual) + bf16 (next GEMM operand).
// ---------------------------------------------------------------------------
__global__ void k_lnu(const float* __restrict__ z, const float* __restrict__ g,
                      const float* __restrict__ b, float* __restrict__ u,
                      __bf16* __restrict__ u16) {
  const int tok = blockIdx.x;
  const float* zr = z + (size_t)tok * HID;
  float s = 0.f, s2 = 0.f;
  for (int i = threadIdx.x; i < HID; i += blockDim.x) {
    const float v = zr[i];
    s += v; s2 += v * v;
  }
  __shared__ float ws1[8], ws2[8];
  const int wid = threadIdx.x >> 5, lane = threadIdx.x & 31;
#pragma unroll
  for (int off = 16; off > 0; off >>= 1) {
    s  += __shfl_down(s, off, 32);
    s2 += __shfl_down(s2, off, 32);
  }
  if (lane == 0) { ws1[wid] = s; ws2[wid] = s2; }
  __syncthreads();
  if (wid == 0) {
    s  = (lane < 8) ? ws1[lane] : 0.f;
    s2 = (lane < 8) ? ws2[lane] : 0.f;
#pragma unroll
    for (int off = 4; off > 0; off >>= 1) {
      s  += __shfl_down(s, off, 32);
      s2 += __shfl_down(s2, off, 32);
    }
    if (lane == 0) { ws1[0] = s; ws2[0] = s2; }
  }
  __syncthreads();
  const float mu  = ws1[0] * (1.f / HID);
  const float var = ws2[0] * (1.f / HID) - mu * mu;
  const float r   = rsqrtf(var + 1e-5f);
  float*  ur   = u   + (size_t)tok * HID;
  __bf16* ur16 = u16 + (size_t)tok * HID;
  for (int i = threadIdx.x; i < HID; i += blockDim.x) {
    const float v = zr[i];
    const float uu = v + (v - mu) * r * g[i] + b[i];
    ur[i]   = uu;
    ur16[i] = (__bf16)uu;
  }
}

// ---------------------------------------------------------------------------
// 7) g16 = bf16(gelu(u @ f1^T + b1))
// ---------------------------------------------------------------------------
__global__ void k_ffn1(const __bf16* __restrict__ u, const __bf16* __restrict__ W,
                       const float* __restrict__ bias, __bf16* __restrict__ g) {
  __shared__ __bf16 sB[2 * BUFELEMS];
  const int ng    = blockIdx.x & 63;          // FF/64 = 64 n-groups
  const int mblk  = blockIdx.x >> 6;
  const int mBase = mblk * 128 + (threadIdx.x >> 5) * 16;
  const int nBase = ng * 64;
  const int lane  = threadIdx.x & 31;

  v8f acc[4] = {};
  coop_gemm(u, HID, W, HID, mBase, nBase, HID, sB, acc);

  const int n = lane & 15, hk = lane >> 4;
#pragma unroll
  for (int t = 0; t < 4; ++t) {
    const int col = nBase + 16 * t + n;
    const float bv = bias[col];
#pragma unroll
    for (int j = 0; j < 8; ++j) {
      const int row = mBase + j + 8 * hk;
      const float x = acc[t][j] + bv;
      g[(size_t)row * FF + col] =
          (__bf16)(0.5f * x * (1.f + erff(x * 0.70710678118654752f)));
    }
  }
}

// ---------------------------------------------------------------------------
// 8) v16 = bf16(u + (g @ f2^T + b2))      (K = 4096)
// ---------------------------------------------------------------------------
__global__ void k_ffn2(const __bf16* __restrict__ g, const __bf16* __restrict__ W,
                       const float* __restrict__ bias, const float* __restrict__ u,
                       __bf16* __restrict__ v) {
  __shared__ __bf16 sB[2 * BUFELEMS];
  const int ng    = blockIdx.x & 15;
  const int mblk  = blockIdx.x >> 4;
  const int mBase = mblk * 128 + (threadIdx.x >> 5) * 16;
  const int nBase = ng * 64;
  const int lane  = threadIdx.x & 31;

  v8f acc[4] = {};
  coop_gemm(g, FF, W, FF, mBase, nBase, FF, sB, acc);

  const int n = lane & 15, hk = lane >> 4;
#pragma unroll
  for (int t = 0; t < 4; ++t) {
    const int col = nBase + 16 * t + n;
    const float bv = bias[col];
#pragma unroll
    for (int j = 0; j < 8; ++j) {
      const int row = mBase + j + 8 * hk;
      const size_t idx = (size_t)row * HID + col;
      v[idx] = (__bf16)(u[idx] + acc[t][j] + bv);
    }
  }
}

// ---------------------------------------------------------------------------
// 9) out = v @ p^T + p_b
// ---------------------------------------------------------------------------
__global__ void k_pout(const __bf16* __restrict__ v, const __bf16* __restrict__ W,
                       const float* __restrict__ bias, float* __restrict__ out) {
  __shared__ __bf16 sB[2 * BUFELEMS];
  const int ng    = blockIdx.x & 15;
  const int mblk  = blockIdx.x >> 4;
  const int mBase = mblk * 128 + (threadIdx.x >> 5) * 16;
  const int nBase = ng * 64;
  const int lane  = threadIdx.x & 31;

  v8f acc[4] = {};
  coop_gemm(v, HID, W, HID, mBase, nBase, HID, sB, acc);

  const int n = lane & 15, hk = lane >> 4;
#pragma unroll
  for (int t = 0; t < 4; ++t) {
    const int col = nBase + 16 * t + n;
    const float bv = bias[col];
#pragma unroll
    for (int j = 0; j < 8; ++j) {
      const int row = mBase + j + 8 * hk;
      out[(size_t)row * HID + col] = acc[t][j] + bv;
    }
  }
}

// ---------------------------------------------------------------------------
extern "C" void kernel_launch(void* const* d_in, const int* in_sizes, int n_in,
                              void* d_out, int out_size, void* d_ws, size_t ws_size,
                              hipStream_t stream) {
  const float* emb  = (const float*)d_in[0];
  const float* Wa_w = (const float*)d_in[1];
  const float* Wa_b = (const float*)d_in[2];
  const float* Wb_w = (const float*)d_in[3];
  const float* Wb_b = (const float*)d_in[4];
  const float* Wo_w = (const float*)d_in[5];
  const float* Wo_b = (const float*)d_in[6];
  const float* C_w  = (const float*)d_in[7];
  const float* C_b  = (const float*)d_in[8];
  const float* ln_g = (const float*)d_in[9];
  const float* ln_b = (const float*)d_in[10];
  const float* f1_w = (const float*)d_in[11];
  const float* f1_b = (const float*)d_in[12];
  const float* f2_w = (const float*)d_in[13];
  const float* f2_b = (const float*)d_in[14];
  const float* p_w  = (const float*)d_in[15];
  const float* p_b  = (const float*)d_in[16];
  const float* hw   = (const float*)d_in[17];
  float* out = (float*)d_out;

  char* wp = (char*)d_ws;
  auto alloc_f = [&](size_t n) { float*  r = (float*)wp;  wp += n * 4; return r; };
  auto alloc_h = [&](size_t n) { __bf16* r = (__bf16*)wp; wp += n * 2; return r; };

  float* abuf = alloc_f((size_t)NTOK * HID);   // a (reused as z after scan)
  float* bbuf = alloc_f((size_t)NTOK * HID);   // b
  float* wbuf = alloc_f((size_t)NTOK * HID);   // w
  float* ubuf = alloc_f((size_t)NTOK * HID);   // u (fp32, residual)
  float* Aprod  = alloc_f((size_t)CHAINS * NCHUNK);
  float* Bfin   = alloc_f((size_t)CHAINS * NCHUNK);
  float* hstart = alloc_f((size_t)CHAINS * NCHUNK);

  __bf16* e16  = alloc_h((size_t)NTOK * EMBED);
  __bf16* h16  = alloc_h((size_t)NTOK * HID);
  __bf16* u16  = alloc_h((size_t)NTOK * HID);
  __bf16* g16  = alloc_h((size_t)NTOK * FF);
  __bf16* v16  = alloc_h((size_t)NTOK * HID);
  __bf16* Wa16 = alloc_h((size_t)HID * EMBED);
  __bf16* Wb16 = alloc_h((size_t)HID * EMBED);
  __bf16* Wo16 = alloc_h((size_t)HID * EMBED);
  __bf16* C16  = alloc_h((size_t)HID * HID);
  __bf16* f116 = alloc_h((size_t)FF * HID);
  __bf16* f216 = alloc_h((size_t)HID * FF);
  __bf16* p16  = alloc_h((size_t)HID * HID);
  float* zbuf = abuf;   // a dead after the scan

  const dim3 blk(256);
  // pack weights + embeddings to bf16 once
  k_pack4<<<(NTOK * EMBED) / 1024, blk, 0, stream>>>(emb, e16, NTOK * EMBED);
  k_pack4<<<(HID * EMBED) / 1024, blk, 0, stream>>>(Wa_w, Wa16, HID * EMBED);
  k_pack4<<<(HID * EMBED) / 1024, blk, 0, stream>>>(Wb_w, Wb16, HID * EMBED);
  k_pack4<<<(HID * EMBED) / 1024, blk, 0, stream>>>(Wo_w, Wo16, HID * EMBED);
  k_pack4<<<(HID * HID) / 1024, blk, 0, stream>>>(C_w, C16, HID * HID);
  k_pack4<<<(FF * HID) / 1024, blk, 0, stream>>>(f1_w, f116, FF * HID);
  k_pack4<<<(HID * FF) / 1024, blk, 0, stream>>>(f2_w, f216, HID * FF);
  k_pack4<<<(HID * HID) / 1024, blk, 0, stream>>>(p_w, p16, HID * HID);

  // 1) a,b,w : 128 m-blocks * 16 n-groups (block = 128x64, 3 outputs)
  k_abw<<<2048, blk, 0, stream>>>(e16, Wa16, Wa_b, Wb16, Wb_b, Wo16, Wo_b,
                                  abuf, bbuf, wbuf);
  // 2-4) scan
  k_scan_local<<<(CHAINS * NCHUNK) / 256, blk, 0, stream>>>(abuf, bbuf, Aprod, Bfin);
  k_scan_carry<<<CHAINS / 256, blk, 0, stream>>>(Aprod, Bfin, hstart);
  k_scan_apply<<<(CHAINS * NCHUNK) / 256, blk, 0, stream>>>(abuf, bbuf, hstart, h16);
  // 5) z : 128 m-blocks * 16 n-groups
  k_cz<<<2048, blk, 0, stream>>>(h16, C16, C_b, wbuf, hw, zbuf);
  // 6) u
  k_lnu<<<NTOK, blk, 0, stream>>>(zbuf, ln_g, ln_b, ubuf, u16);
  // 7) ffn hidden : 128 * 64
  k_ffn1<<<8192, blk, 0, stream>>>(u16, f116, f1_b, g16);
  // 8) v = u + ffn
  k_ffn2<<<2048, blk, 0, stream>>>(g16, f216, f2_b, ubuf, v16);
  // 9) out
  k_pout<<<2048, blk, 0, stream>>>(v16, p16, p_b, out);
}
